// GCN_26877905339050
// MI455X (gfx1250) — matrix operand order repaired and checked by reading, not compiled, not run on hardware
//
#include <hip/hip_runtime.h>
#include <hip/hip_bf16.h>
#include <stdint.h>

typedef __attribute__((ext_vector_type(16))) _Float16 v16h;
typedef __attribute__((ext_vector_type(8)))  _Float16 v8h;
typedef __attribute__((ext_vector_type(8)))  float    v8f;

#define CDIV(a,b) (((a)+(b)-1)/(b))

// ---------------------------------------------------------------------------
// Small elementwise kernels
// ---------------------------------------------------------------------------

// h = x @ node_W.T + node_b   (N x 7) -> (N x 128), write f32 + f16 copies
__global__ void node_embed_kernel(const float* __restrict__ x,
                                  const float* __restrict__ node_W,
                                  const float* __restrict__ node_b,
                                  float* __restrict__ h32,
                                  _Float16* __restrict__ h16,
                                  int N) {
    int idx = blockIdx.x * blockDim.x + threadIdx.x;
    if (idx >= N * 128) return;
    int node = idx >> 7;
    int c    = idx & 127;
    float s = node_b[c];
    const float* xr = x + (size_t)node * 7;
    const float* wr = node_W + (size_t)c * 7;
    #pragma unroll
    for (int k = 0; k < 7; ++k) s += xr[k] * wr[k];
    h32[idx] = s;
    h16[idx] = (_Float16)s;
}

// e = edge_attr @ edge_W.T + edge_b   (E x 6) -> (E x 128) f16
__global__ void edge_embed_kernel(const float* __restrict__ ea,
                                  const float* __restrict__ edge_W,
                                  const float* __restrict__ edge_b,
                                  _Float16* __restrict__ e16,
                                  int E) {
    long long idx = (long long)blockIdx.x * blockDim.x + threadIdx.x;
    if (idx >= (long long)E * 128) return;
    long long edge = idx >> 7;
    int c = (int)(idx & 127);
    float s = edge_b[c];
    const float* ar = ea + edge * 6;
    const float* wr = edge_W + (size_t)c * 6;
    #pragma unroll
    for (int k = 0; k < 6; ++k) s += ar[k] * wr[k];
    e16[idx] = (_Float16)s;
}

// Pack Wf/Ws into f16 weight matrices:
//   W4h[l] (512 x 128): rows 0..127=Wfi, 128..255=Wfj, 256..383=Wsi, 384..511=Wsj
//   WEh[l] (256 x 128): rows 0..127=Wfe, 128..255=Wse
__global__ void pack_weights_kernel(const float* __restrict__ Wf,
                                    const float* __restrict__ Ws,
                                    _Float16* __restrict__ W4h,
                                    _Float16* __restrict__ WEh,
                                    int L) {
    int n4 = L * 512 * 128;
    int nE = L * 256 * 128;
    int idx = blockIdx.x * blockDim.x + threadIdx.x;
    if (idx < n4) {
        int l   = idx / (512 * 128);
        int rem = idx % (512 * 128);
        int r   = rem >> 7;
        int k   = rem & 127;
        float v;
        if (r < 128)       v = Wf[((size_t)l * 128 + r)       * 384 + k];
        else if (r < 256)  v = Wf[((size_t)l * 128 + (r-128)) * 384 + 128 + k];
        else if (r < 384)  v = Ws[((size_t)l * 128 + (r-256)) * 384 + k];
        else               v = Ws[((size_t)l * 128 + (r-384)) * 384 + 128 + k];
        W4h[idx] = (_Float16)v;
    } else if (idx < n4 + nE) {
        int i2  = idx - n4;
        int l   = i2 / (256 * 128);
        int rem = i2 % (256 * 128);
        int r   = rem >> 7;
        int k   = rem & 127;
        float v = (r < 128)
            ? Wf[((size_t)l * 128 + r)       * 384 + 256 + k]
            : Ws[((size_t)l * 128 + (r-128)) * 384 + 256 + k];
        WEh[i2] = (_Float16)v;
    }
}

__global__ void zero_agg_kernel(unsigned int* __restrict__ agg, int n) {
    int idx = blockIdx.x * blockDim.x + threadIdx.x;
    if (idx < n) agg[idx] = 0u;   // 0u == bits of +0.0f; msg >= 0 so this is the identity
}

// h += BN(agg);  refresh h16
__global__ void update_h_kernel(const unsigned int* __restrict__ agg,
                                const float* __restrict__ gamma,
                                const float* __restrict__ beta,
                                const float* __restrict__ mean,
                                const float* __restrict__ var,
                                float* __restrict__ h32,
                                _Float16* __restrict__ h16,
                                int N) {
    int idx = blockIdx.x * blockDim.x + threadIdx.x;
    if (idx >= N * 128) return;
    int c = idx & 127;
    float a = __uint_as_float(agg[idx]);
    float norm = (a - mean[c]) * rsqrtf(var[c] + 1e-5f) * gamma[c] + beta[c];
    float hn = h32[idx] + norm;
    h32[idx] = hn;
    h16[idx] = (_Float16)hn;
}

__global__ void copy_h_kernel(const float* __restrict__ h32,
                              float* __restrict__ out_h, int n) {
    int idx = blockIdx.x * blockDim.x + threadIdx.x;
    if (idx < n) out_h[idx] = h32[idx];
}

// out = h @ lin_W.T + lin_b  (K = 10)
__global__ void lin_kernel(const float* __restrict__ h32,
                           const float* __restrict__ lin_W,
                           const float* __restrict__ lin_b,
                           float* __restrict__ out, int N) {
    int node = blockIdx.x * blockDim.x + threadIdx.x;
    if (node >= N) return;
    float acc[10];
    #pragma unroll
    for (int k = 0; k < 10; ++k) acc[k] = lin_b[k];
    const float* hr = h32 + (size_t)node * 128;
    for (int c = 0; c < 128; ++c) {
        float hv = hr[c];
        #pragma unroll
        for (int k = 0; k < 10; ++k) acc[k] += hv * lin_W[k * 128 + c];
    }
    #pragma unroll
    for (int k = 0; k < 10; ++k) out[(size_t)node * 10 + k] = acc[k];
}

// ---------------------------------------------------------------------------
// WMMA helpers
// A (16x32 f16) fragment: lane half=l>>4, m=l&15:
//   elems 0..7  = act[m][k0 + 16*half + j]
//   elems 8..15 = act[m][k0 + 16*half + 8 + j]
// B (32x16 f16) fragment: lane n=l&15, half=l>>4:
//   elems 0..15 = W[n0+n][k0 + 16*half + j]  (contiguous 32B load!)
// C/D (16x16 f32): acc[v] = element (m = v + 8*half, n = l&15)
// ---------------------------------------------------------------------------

__device__ __forceinline__ void load_a_frags(const _Float16* __restrict__ act,
                                             int row0, int n, int half,
                                             v16h a[4]) {
    const _Float16* arow = act + (size_t)(row0 + n) * 128;
    #pragma unroll
    for (int kk = 0; kk < 4; ++kk) {
        v8h lo = *(const v8h*)(arow + kk * 32 + half * 16);
        v8h hi = *(const v8h*)(arow + kk * 32 + half * 16 + 8);
        a[kk] = __builtin_shufflevector(lo, hi, 0,1,2,3,4,5,6,7,8,9,10,11,12,13,14,15);
    }
}

// ---------------------------------------------------------------------------
// Node GEMM: H4 = h @ [Wfi;Wfj;Wsi;Wsj].T  -> (N x 512) f32
// One wave per 16-node tile; 32 output N-tiles of 16.
// ---------------------------------------------------------------------------
__global__ void node_gemm_kernel(const _Float16* __restrict__ h16,
                                 const _Float16* __restrict__ W4l,
                                 float* __restrict__ H4, int N) {
    int wave = (blockIdx.x * blockDim.x + threadIdx.x) >> 5;
    int lane = threadIdx.x & 31;
    int r0 = wave * 16;
    if (r0 >= N) return;                    // wave-uniform branch (EXEC stays all-1)
    int n = lane & 15, half = lane >> 4;

    v16h a[4];
    load_a_frags(h16, r0, n, half, a);

    for (int j = 0; j < 32; ++j) {
        v8f acc = {};
        const _Float16* wrow = W4l + (size_t)(j * 16 + n) * 128 + half * 16;
        #pragma unroll
        for (int kk = 0; kk < 4; ++kk) {
            v16h b = *(const v16h*)(wrow + kk * 32);
            acc = __builtin_amdgcn_wmma_f32_16x16x32_f16(
                false, a[kk], false, b, (short)0, acc, false, false);
        }
        #pragma unroll
        for (int v = 0; v < 8; ++v) {
            int m = v + 8 * half;
            H4[(size_t)(r0 + m) * 512 + j * 16 + n] = acc[v];
        }
    }
}

// ---------------------------------------------------------------------------
// Fused edge kernel: zf/zs = e@[Wfe;Wse].T + gather(H4) + bias,
// msg = sigmoid(zf)*softplus(zs), segment-max via u32 atomicMax (msg >= 0).
// One wave per 16-edge tile.
// ---------------------------------------------------------------------------
__global__ void edge_msg_kernel(const _Float16* __restrict__ e16,
                                const float* __restrict__ H4,
                                const _Float16* __restrict__ WEl,
                                const float* __restrict__ bfl,
                                const float* __restrict__ bsl,
                                const long long* __restrict__ src,
                                const long long* __restrict__ dst,
                                unsigned int* __restrict__ agg,
                                int E) {
    int wave = (blockIdx.x * blockDim.x + threadIdx.x) >> 5;
    int lane = threadIdx.x & 31;
    int e0 = wave * 16;
    if (e0 >= E) return;                    // wave-uniform
    int n = lane & 15, half = lane >> 4;

    v16h a[4];
    load_a_frags(e16, e0, n, half, a);

    long long dIdx[8], sIdx[8];
    #pragma unroll
    for (int v = 0; v < 8; ++v) {
        int m = v + 8 * half;
        dIdx[v] = dst[e0 + m];
        sIdx[v] = src[e0 + m];
    }

    for (int j = 0; j < 8; ++j) {
        v8f accf = {};
        v8f accs = {};
        const _Float16* wf_row = WEl + (size_t)(j * 16 + n)       * 128 + half * 16;
        const _Float16* ws_row = WEl + (size_t)(128 + j * 16 + n) * 128 + half * 16;
        #pragma unroll
        for (int kk = 0; kk < 4; ++kk) {
            v16h bfr = *(const v16h*)(wf_row + kk * 32);
            v16h bsr = *(const v16h*)(ws_row + kk * 32);
            accf = __builtin_amdgcn_wmma_f32_16x16x32_f16(
                false, a[kk], false, bfr, (short)0, accf, false, false);
            accs = __builtin_amdgcn_wmma_f32_16x16x32_f16(
                false, a[kk], false, bsr, (short)0, accs, false, false);
        }
        int ch = j * 16 + n;
        float bfv = bfl[ch], bsv = bsl[ch];
        #pragma unroll
        for (int v = 0; v < 8; ++v) {
            const float* Hd = H4 + (size_t)dIdx[v] * 512;
            const float* Hs = H4 + (size_t)sIdx[v] * 512;
            float zf = accf[v] + Hd[ch]       + Hs[128 + ch] + bfv;
            float zs = accs[v] + Hd[256 + ch] + Hs[384 + ch] + bsv;
            float sig = 1.0f / (1.0f + __expf(-zf));
            float sp  = (zs > 20.0f) ? zs : log1pf(__expf(zs));
            float msg = sig * sp;                        // >= 0
            atomicMax(agg + (size_t)dIdx[v] * 128 + ch, __float_as_uint(msg));
        }
    }
}

// ---------------------------------------------------------------------------
// Host launcher
// ---------------------------------------------------------------------------
extern "C" void kernel_launch(void* const* d_in, const int* in_sizes, int n_in,
                              void* d_out, int out_size, void* d_ws, size_t ws_size,
                              hipStream_t stream) {
    const float*     x       = (const float*)d_in[0];
    const long long* ei      = (const long long*)d_in[1];
    const float*     ea      = (const float*)d_in[2];
    const float*     node_W  = (const float*)d_in[3];
    const float*     node_b  = (const float*)d_in[4];
    const float*     edge_W  = (const float*)d_in[5];
    const float*     edge_b  = (const float*)d_in[6];
    const float*     Wf      = (const float*)d_in[7];
    const float*     bf      = (const float*)d_in[8];
    const float*     Ws      = (const float*)d_in[9];
    const float*     bs      = (const float*)d_in[10];
    const float*     gamma   = (const float*)d_in[11];
    const float*     beta    = (const float*)d_in[12];
    const float*     bn_mean = (const float*)d_in[13];
    const float*     bn_var  = (const float*)d_in[14];
    const float*     lin_W   = (const float*)d_in[15];
    const float*     lin_b   = (const float*)d_in[16];

    const int N = in_sizes[0] / 7;       // 50000
    const int E = in_sizes[2] / 6;       // 800000
    const int L = in_sizes[8] / 128;     // 3
    const int K = in_sizes[16];          // 10

    // Workspace layout (all offsets 256B-aligned)
    char* ws = (char*)d_ws;
    size_t off = 0;
    float*        h32 = (float*)(ws + off);        off += (size_t)N * 128 * 4;
    _Float16*     h16 = (_Float16*)(ws + off);     off += (size_t)N * 128 * 2;
    _Float16*     e16 = (_Float16*)(ws + off);     off += (size_t)E * 128 * 2;
    float*        H4  = (float*)(ws + off);        off += (size_t)N * 512 * 4;
    unsigned int* agg = (unsigned int*)(ws + off); off += (size_t)N * 128 * 4;
    _Float16*     W4h = (_Float16*)(ws + off);     off += (size_t)L * 512 * 128 * 2;
    _Float16*     WEh = (_Float16*)(ws + off);     off += (size_t)L * 256 * 128 * 2;
    (void)off; (void)ws_size; (void)n_in;

    const long long* src = ei;       // edge_index[0]
    const long long* dst = ei + E;   // edge_index[1]

    const int TB = 256;

    // Prep
    {
        int tot = L * (512 + 256) * 128;
        pack_weights_kernel<<<CDIV(tot, TB), TB, 0, stream>>>(Wf, Ws, W4h, WEh, L);
    }
    node_embed_kernel<<<CDIV(N * 128, TB), TB, 0, stream>>>(x, node_W, node_b, h32, h16, N);
    {
        long long tot = (long long)E * 128;
        edge_embed_kernel<<<(int)CDIV(tot, TB), TB, 0, stream>>>(ea, edge_W, edge_b, e16, E);
    }

    const int nodeWaves = CDIV(N, 16);
    const int edgeWaves = CDIV(E, 16);

    for (int l = 0; l < L; ++l) {
        node_gemm_kernel<<<CDIV(nodeWaves, 8), TB, 0, stream>>>(
            h16, W4h + (size_t)l * 512 * 128, H4, N);
        zero_agg_kernel<<<CDIV(N * 128, TB), TB, 0, stream>>>(agg, N * 128);
        edge_msg_kernel<<<CDIV(edgeWaves, 8), TB, 0, stream>>>(
            e16, H4, WEh + (size_t)l * 256 * 128,
            bf + (size_t)l * 128, bs + (size_t)l * 128,
            src, dst, agg, E);
        update_h_kernel<<<CDIV(N * 128, TB), TB, 0, stream>>>(
            agg, gamma + (size_t)l * 128, beta + (size_t)l * 128,
            bn_mean + (size_t)l * 128, bn_var + (size_t)l * 128,
            h32, h16, N);
    }

    float* out_logits = (float*)d_out;            // (N, K)
    float* out_h      = (float*)d_out + (size_t)N * K;  // (N, C)
    lin_kernel<<<CDIV(N, TB), TB, 0, stream>>>(h32, lin_W, lin_b, out_logits, N);
    copy_h_kernel<<<CDIV(N * 128, TB), TB, 0, stream>>>(h32, out_h, N * 128);
}
